// MusicFiLMLayer_61761629716611
// MI455X (gfx1250) — matrix operand order repaired and checked by reading, not compiled, not run on hardware
//
#include <hip/hip_runtime.h>
#include <hip/hip_bf16.h>

// ---------------------------------------------------------------------------
// MusicFiLM layer for MI455X (gfx1250): WMMA bf16 GEMM pipeline + fused
// selective-scan. wave32 everywhere; 256-thread blocks = 8 waves.
// GEMM tiles are staged with CDNA5 async global->LDS DMA (ASYNCcnt).
// ---------------------------------------------------------------------------

typedef __attribute__((ext_vector_type(16))) __bf16 bf16x16;
typedef __attribute__((ext_vector_type(8)))  float  floatx8;

#define BM 64
#define BN 64
#define BK 32

// ----------------------------- helpers -------------------------------------

__device__ __forceinline__ unsigned short f2bf(float f) {
    unsigned int u = __float_as_uint(f);
    unsigned int r = u + 0x7FFFu + ((u >> 16) & 1u);   // round-to-nearest-even
    return (unsigned short)(r >> 16);
}
__device__ __forceinline__ float bf2f(unsigned short h) {
    return __uint_as_float(((unsigned int)h) << 16);
}
__device__ __forceinline__ float sigmoidf_(float x) {
    return 1.0f / (1.0f + __expf(-x));
}
__device__ __forceinline__ float siluf_(float x) {
    return x * sigmoidf_(x);
}
__device__ __forceinline__ float gelu_exact(float x) {
    return 0.5f * x * (1.0f + erff(x * 0.70710678118654752f));
}
__device__ __forceinline__ float softplusf_(float x) {
    return (x > 20.0f) ? x : log1pf(__expf(x));
}

// wave-relative LDS byte offset of a __shared__ object (flat addr low bits)
__device__ __forceinline__ unsigned lds_off(const void* p) {
    return (unsigned)(size_t)p;
}

// CDNA5 async DMA: global -> LDS, 16 bytes per lane, tracked by ASYNCcnt.
__device__ __forceinline__ void async_copy_b128(unsigned lds_dst, const void* gsrc) {
    asm volatile("global_load_async_to_lds_b128 %0, %1, off"
                 :: "v"(lds_dst), "v"(gsrc) : "memory");
}
__device__ __forceinline__ void wait_async0() {
    asm volatile("s_wait_asynccnt 0x0" ::: "memory");
}

// ----------------------------- small kernels --------------------------------

__global__ __launch_bounds__(256) void f32_to_bf16_kernel(
    const float* __restrict__ s, unsigned short* __restrict__ d, int n) {
    int i = blockIdx.x * 256 + threadIdx.x;
    if (i < n) d[i] = f2bf(s[i]);
}

// g[b] = emb[genre[b]] @ pg_w.T + pg_b          (B=8, D=512)
__global__ __launch_bounds__(256) void genre_g_kernel(
    const int* __restrict__ genre, const float* __restrict__ emb,
    const float* __restrict__ pgw, const float* __restrict__ pgb,
    float* __restrict__ g) {
    __shared__ float e[512];
    const int bb = blockIdx.x, tid = threadIdx.x;
    const int gi = genre[bb];
    e[tid]       = emb[gi * 512 + tid];
    e[tid + 256] = emb[gi * 512 + 256 + tid];
    __syncthreads();
    for (int j = tid; j < 512; j += 256) {
        float acc = pgb[j];
        const float* wr = pgw + (size_t)j * 512;
        for (int k = 0; k < 512; ++k) acc += e[k] * wr[k];
        g[bb * 512 + j] = acc;
    }
}

// z(b,l,:) = [ x(b,l,:) , g(b,:) ]  -> bf16 (M x 1024)
__global__ __launch_bounds__(256) void z_build_kernel(
    const float* __restrict__ x, const float* __restrict__ g,
    unsigned short* __restrict__ zh, int Ls) {
    size_t idx = (size_t)blockIdx.x * 256 + threadIdx.x;
    int col = (int)(idx & 1023);
    size_t row = idx >> 10;
    float v;
    if (col < 512) v = x[row * 512 + col];
    else           v = g[(row / (size_t)Ls) * 512 + (col - 512)];
    zh[idx] = f2bf(v);
}

// LayerNorm over D=512; writes forward and L-flipped bf16 copies.
__global__ __launch_bounds__(256) void layernorm_kernel(
    const float* __restrict__ xg, const float* __restrict__ wln,
    const float* __restrict__ bln, unsigned short* __restrict__ xn,
    unsigned short* __restrict__ xnf, int Ls) {
    __shared__ float red[256];
    const int row = blockIdx.x, tid = threadIdx.x;
    const float v0 = xg[(size_t)row * 512 + tid];
    const float v1 = xg[(size_t)row * 512 + 256 + tid];
    red[tid] = v0 + v1;
    __syncthreads();
    for (int s = 128; s > 0; s >>= 1) { if (tid < s) red[tid] += red[tid + s]; __syncthreads(); }
    const float mean = red[0] * (1.0f / 512.0f);
    __syncthreads();
    red[tid] = v0 * v0 + v1 * v1;
    __syncthreads();
    for (int s = 128; s > 0; s >>= 1) { if (tid < s) red[tid] += red[tid + s]; __syncthreads(); }
    const float var  = red[0] * (1.0f / 512.0f) - mean * mean;
    const float rstd = rsqrtf(var + 1e-5f);
    const float o0 = (v0 - mean) * rstd * wln[tid]       + bln[tid];
    const float o1 = (v1 - mean) * rstd * wln[tid + 256] + bln[tid + 256];
    const int b = row / Ls, l = row % Ls;
    const size_t fr = (size_t)b * Ls + (Ls - 1 - l);
    xn [(size_t)row * 512 + tid]        = f2bf(o0);
    xn [(size_t)row * 512 + 256 + tid]  = f2bf(o1);
    xnf[fr * 512 + tid]                 = f2bf(o0);
    xnf[fr * 512 + 256 + tid]           = f2bf(o1);
}

// Causal depthwise conv (K=4) + bias + SiLU on xc half; SiLU on z half.
__global__ __launch_bounds__(256) void conv_silu_kernel(
    const float* __restrict__ xz, const float* __restrict__ cw,
    const float* __restrict__ cb, unsigned short* __restrict__ xch,
    unsigned short* __restrict__ szh, int Ls) {
    size_t idx = (size_t)blockIdx.x * 256 + threadIdx.x;   // M*1024 threads
    const int d = (int)(idx & 1023);
    const size_t row = idx >> 10;
    const int l = (int)(row & (size_t)(Ls - 1));
    float acc = cb[d];
#pragma unroll
    for (int k = 0; k < 4; ++k) {
        const int ll = l - 3 + k;
        if (ll >= 0) acc += cw[d * 4 + k] * xz[(row - 3 + (size_t)k) * 2048 + d];
    }
    xch[idx] = f2bf(siluf_(acc));
    const float z = xz[row * 2048 + 1024 + d];
    szh[idx] = f2bf(siluf_(z));
}

// Sequential selective scan: one thread per (b, d); 16 SSM states in regs.
// y = scan(...) + xc*D, then y *= silu(z), written bf16 for out_proj GEMM.
__global__ __launch_bounds__(256) void ssm_scan_kernel(
    const float* __restrict__ dtf,            // M x 1024 (softplus'd)
    const float* __restrict__ xdbf,           // M x 64: [dt_raw(32) | B(16) | C(16)]
    const unsigned short* __restrict__ xch,   // M x 1024 bf16 (conv+silu)
    const unsigned short* __restrict__ szh,   // M x 1024 bf16 silu(z)
    const float* __restrict__ A_log,          // 1024 x 16
    const float* __restrict__ Dvec,           // 1024
    unsigned short* __restrict__ yh,          // M x 1024 bf16
    int Ls) {
    __shared__ float sBC[32];
    const int tid = threadIdx.x;
    const int b = blockIdx.x >> 2;
    const int d = ((blockIdx.x & 3) << 8) + tid;
    float Av[16], h[16];
#pragma unroll
    for (int n = 0; n < 16; ++n) { Av[n] = -__expf(A_log[d * 16 + n]); h[n] = 0.0f; }
    const float Dv = Dvec[d];
    for (int l = 0; l < Ls; ++l) {
        const size_t row = (size_t)b * Ls + l;
        __syncthreads();
        if (tid < 32) sBC[tid] = xdbf[row * 64 + 32 + tid];
        __syncthreads();
        const float dtv = dtf[row * 1024 + d];
        const float xcv = bf2f(xch[row * 1024 + d]);
        float y = 0.0f;
#pragma unroll
        for (int n = 0; n < 16; ++n) {
            h[n] = __expf(dtv * Av[n]) * h[n] + dtv * sBC[n] * xcv;
            y += h[n] * sBC[16 + n];
        }
        y += xcv * Dv;
        const float zz = bf2f(szh[row * 1024 + d]);
        yh[row * 1024 + d] = f2bf(y * zz);
    }
}

// ----------------------------- WMMA GEMM ------------------------------------
// C(M x N) = epi( A(M x K, bf16, row stride lda) * W(N x K, bf16)^T + bias )
// epi: 0 f32 store | 1 bf16(gelu) | 2 aux0 + sigmoid(v)*aux1 -> f32
//      3 f32 softplus | 4 f32 store w/ optional L-flip | 5 f32 += w/ flip
//      6 f32 store AND bf16 store

__device__ __forceinline__ void gemm_epilogue(
    int epi, int flip, int Ls, int Nc, int r, int n, float v,
    float* __restrict__ outf, unsigned short* __restrict__ outh,
    const float* __restrict__ aux0, const float* __restrict__ aux1) {
    const size_t idx = (size_t)r * Nc + n;
    switch (epi) {
        case 0: outf[idx] = v; break;
        case 1: outh[idx] = f2bf(gelu_exact(v)); break;
        case 2: outf[idx] = aux0[idx] + sigmoidf_(v) * aux1[idx]; break;
        case 3: outf[idx] = softplusf_(v); break;
        case 4:
        case 5: {
            const int bidx = r / Ls, l = r % Ls;
            const int r2 = flip ? (bidx * Ls + (Ls - 1 - l)) : r;
            const size_t i2 = (size_t)r2 * Nc + n;
            if (epi == 4) outf[i2] = v; else outf[i2] += v;
        } break;
        case 6: outf[idx] = v; outh[idx] = f2bf(v); break;
    }
}

__global__ __launch_bounds__(256) void gemm_bf16_wmma(
    const unsigned short* __restrict__ A, int lda,
    const unsigned short* __restrict__ W,      // N x K row-major
    const float* __restrict__ bias,            // length N or nullptr
    int Nc, int Kk, int epi, int flip, int Ls,
    float* __restrict__ outf, unsigned short* __restrict__ outh,
    const float* __restrict__ aux0, const float* __restrict__ aux1) {
    __shared__ unsigned short As[2][BM][BK + 8];
    __shared__ unsigned short Ws[2][BN][BK + 8];

    const int tid  = threadIdx.x;
    const int lane = tid & 31, wave = tid >> 5;
    const int rg = wave & 3, cg = wave >> 2;
    const int by = blockIdx.y, bx = blockIdx.x;

    // staging: each of 256 threads DMAs one 16B chunk per tile per matrix
    const int lr = tid >> 2;            // 0..63
    const int lc = (tid & 3) << 3;      // 0,8,16,24
    const size_t aOff = (size_t)(by * BM + lr) * lda + lc;
    const size_t wOff = (size_t)(bx * BN + lr) * Kk  + lc;

    const int nk = Kk / BK;

    auto loadTile = [&](int kt, int buf) {
        if (kt + 2 < nk) {  // WGP-scope prefetch of tile kt+2 -> global_prefetch_b8
            __builtin_prefetch((const void*)(A + aOff + (size_t)(kt + 2) * BK), 0, 3);
            __builtin_prefetch((const void*)(W + wOff + (size_t)(kt + 2) * BK), 0, 3);
        }
        async_copy_b128(lds_off(&As[buf][lr][lc]),
                        (const void*)(A + aOff + (size_t)kt * BK));
        async_copy_b128(lds_off(&Ws[buf][lr][lc]),
                        (const void*)(W + wOff + (size_t)kt * BK));
    };

    floatx8 acc0, acc1;
#pragma unroll
    for (int i = 0; i < 8; ++i) { acc0[i] = 0.0f; acc1[i] = 0.0f; }

    loadTile(0, 0);

    // Fragment lane mapping used by this code (wave32; contiguous k, same order for A and B):
    // A 16x32 bf16: lane m(0..15) rows, Ks 0..15; lanes 16..31: Ks 16..31
    // B 32x16 bf16: lanes 0..15 hold K=0..15 of col n; lanes 16..31 hold K=16..31
    const int arow  = rg * 16 + (lane & 15);
    const int ka    = (lane >> 4) << 4;    // 0 | 16
    const int kb    = (lane >> 4) << 4;    // 0 | 16
    const int wcol0 = cg * 32 + (lane & 15);
    const int wcol1 = wcol0 + 16;

    for (int kt = 0; kt < nk; ++kt) {
        const int buf = kt & 1;
        wait_async0();        // our async DMAs into `buf` have landed in LDS
        __syncthreads();      // everyone's have
        if (kt + 1 < nk) loadTile(kt + 1, buf ^ 1);   // overlap DMA with math

        union { bf16x16 v; uint4 q[2]; } fa, fb0, fb1;
        fa.q[0]  = *(const uint4*)&As[buf][arow][ka];
        fa.q[1]  = *(const uint4*)&As[buf][arow][ka + 8];
        fb0.q[0] = *(const uint4*)&Ws[buf][wcol0][kb];
        fb0.q[1] = *(const uint4*)&Ws[buf][wcol0][kb + 8];
        fb1.q[0] = *(const uint4*)&Ws[buf][wcol1][kb];
        fb1.q[1] = *(const uint4*)&Ws[buf][wcol1][kb + 8];

        acc0 = __builtin_amdgcn_wmma_f32_16x16x32_bf16(
            false, fa.v, false, fb0.v, (short)0, acc0, false, false);
        acc1 = __builtin_amdgcn_wmma_f32_16x16x32_bf16(
            false, fa.v, false, fb1.v, (short)0, acc1, false, false);
    }

    // D layout: lanes<16: M=j, N=lane; lanes>=16: M=8+j, N=lane-16
    const int mrow0 = by * BM + rg * 16 + ((lane >> 4) << 3);
    const int n0 = bx * BN + cg * 32 + (lane & 15);
    const int n1 = n0 + 16;
    const float bb0 = bias ? bias[n0] : 0.0f;
    const float bb1 = bias ? bias[n1] : 0.0f;
#pragma unroll
    for (int j = 0; j < 8; ++j) {
        const int r = mrow0 + j;
        gemm_epilogue(epi, flip, Ls, Nc, r, n0, acc0[j] + bb0, outf, outh, aux0, aux1);
        gemm_epilogue(epi, flip, Ls, Nc, r, n1, acc1[j] + bb1, outf, outh, aux0, aux1);
    }
}

// ----------------------------- launcher -------------------------------------

extern "C" void kernel_launch(void* const* d_in, const int* in_sizes, int n_in,
                              void* d_out, int out_size, void* d_ws, size_t ws_size,
                              hipStream_t stream) {
    (void)in_sizes; (void)n_in; (void)out_size; (void)ws_size;

    const int Bb = 8, Ls = 2048, Dm = 512, Di = 1024;
    const size_t Mr = (size_t)Bb * Ls;          // 16384

    // ---- inputs (setup_inputs dict order, nested dicts flattened) ----
    const float* x      = (const float*)d_in[0];
    const int*   genre  = (const int*)  d_in[1];
    const float* emb    = (const float*)d_in[2];
    const float* pg_w   = (const float*)d_in[3];
    const float* pg_b   = (const float*)d_in[4];
    const float* d1_w   = (const float*)d_in[5];
    const float* d1_b   = (const float*)d_in[6];
    const float* d2_w   = (const float*)d_in[7];
    const float* d2_b   = (const float*)d_in[8];
    const float* a1_w   = (const float*)d_in[9];
    const float* a1_b   = (const float*)d_in[10];
    const float* a2_w   = (const float*)d_in[11];
    const float* a2_b   = (const float*)d_in[12];
    const float* ln_w   = (const float*)d_in[13];
    const float* ln_b   = (const float*)d_in[14];
    const float* inpw[2]  = { (const float*)d_in[15], (const float*)d_in[24] };
    const float* convw[2] = { (const float*)d_in[16], (const float*)d_in[25] };
    const float* convb[2] = { (const float*)d_in[17], (const float*)d_in[26] };
    const float* xpw[2]   = { (const float*)d_in[18], (const float*)d_in[27] };
    const float* dtpw[2]  = { (const float*)d_in[19], (const float*)d_in[28] };
    const float* dtpb[2]  = { (const float*)d_in[20], (const float*)d_in[29] };
    const float* Alog[2]  = { (const float*)d_in[21], (const float*)d_in[30] };
    const float* Dv[2]    = { (const float*)d_in[22], (const float*)d_in[31] };
    const float* outpw[2] = { (const float*)d_in[23], (const float*)d_in[32] };
    float* out = (float*)d_out;

    // ---- workspace layout ----
    char* wsb = (char*)d_ws;
    size_t off = 0;
    auto alloc = [&](size_t bytes) -> char* {
        char* p = wsb + off; off += (bytes + 255) & ~(size_t)255; return p;
    };
    unsigned short* d1h  = (unsigned short*)alloc((size_t)Dm * 2 * Dm * 2);
    unsigned short* a1h  = (unsigned short*)alloc((size_t)Dm * 2 * Dm * 2);
    unsigned short* d2h  = (unsigned short*)alloc((size_t)Dm * Dm * 2);
    unsigned short* a2h  = (unsigned short*)alloc((size_t)Dm * Dm * 2);
    unsigned short* inph[2], *xph[2], *dtph[2], *outph[2];
    for (int d = 0; d < 2; ++d) {
        inph[d]  = (unsigned short*)alloc((size_t)2 * Di * Dm * 2);
        xph[d]   = (unsigned short*)alloc((size_t)64 * Di * 2);
        dtph[d]  = (unsigned short*)alloc((size_t)Di * 32 * 2);
        outph[d] = (unsigned short*)alloc((size_t)Dm * Di * 2);
    }
    float* gb = (float*)alloc((size_t)Bb * Dm * 4);
    unsigned short* xnh  = (unsigned short*)alloc(Mr * Dm * 2);
    unsigned short* xnfh = (unsigned short*)alloc(Mr * Dm * 2);

    // union region: gate scratch then (re-used) mamba scratch
    const size_t ubase = off;
    size_t uo = ubase;
    auto ualloc = [&](size_t bytes) -> char* {
        char* p = wsb + uo; uo += (bytes + 255) & ~(size_t)255; return p;
    };
    // gate stage
    unsigned short* zh     = (unsigned short*)ualloc(Mr * 1024 * 2);
    unsigned short* h1h    = (unsigned short*)ualloc(Mr * Dm * 2);
    unsigned short* h2h    = (unsigned short*)ualloc(Mr * Dm * 2);
    float*          deltaf = (float*)ualloc(Mr * Dm * 4);
    float*          xgf    = (float*)ualloc(Mr * Dm * 4);
    // mamba stage (overlaps gate stage; strictly later in stream order)
    uo = ubase;
    float*          xzf  = (float*)ualloc(Mr * 2048 * 4);
    unsigned short* xch  = (unsigned short*)ualloc(Mr * Di * 2);
    unsigned short* szh  = (unsigned short*)ualloc(Mr * Di * 2);
    float*          xdbf = (float*)ualloc(Mr * 64 * 4);
    unsigned short* xdbh = (unsigned short*)ualloc(Mr * 64 * 2);
    float*          dtf  = (float*)ualloc(Mr * Di * 4);
    unsigned short* ybf  = (unsigned short*)ualloc(Mr * Di * 2);

    // ---- weight conversion to bf16 ----
    auto cvt = [&](const float* s, unsigned short* d, int n) {
        f32_to_bf16_kernel<<<(n + 255) / 256, 256, 0, stream>>>(s, d, n);
    };
    cvt(d1_w, d1h, Dm * 2 * Dm);
    cvt(a1_w, a1h, Dm * 2 * Dm);
    cvt(d2_w, d2h, Dm * Dm);
    cvt(a2_w, a2h, Dm * Dm);
    for (int d = 0; d < 2; ++d) {
        cvt(inpw[d],  inph[d],  2 * Di * Dm);
        cvt(xpw[d],   xph[d],   64 * Di);
        cvt(dtpw[d],  dtph[d],  Di * 32);
        cvt(outpw[d], outph[d], Dm * Di);
    }

    // ---- genre gate ----
    genre_g_kernel<<<Bb, 256, 0, stream>>>(genre, emb, pg_w, pg_b, gb);
    z_build_kernel<<<(int)(Mr * 1024 / 256), 256, 0, stream>>>(x, gb, zh, Ls);

    const dim3 blk(256);
    // h1 = gelu(z @ d1^T + d1_b)  -> bf16
    gemm_bf16_wmma<<<dim3(Dm / BN, (int)(Mr / BM)), blk, 0, stream>>>(
        zh, 2 * Dm, d1h, d1_b, Dm, 2 * Dm, 1, 0, Ls, nullptr, h1h, nullptr, nullptr);
    // h2 = gelu(z @ a1^T + a1_b)  -> bf16
    gemm_bf16_wmma<<<dim3(Dm / BN, (int)(Mr / BM)), blk, 0, stream>>>(
        zh, 2 * Dm, a1h, a1_b, Dm, 2 * Dm, 1, 0, Ls, nullptr, h2h, nullptr, nullptr);
    // delta = h1 @ d2^T + d2_b -> f32
    gemm_bf16_wmma<<<dim3(Dm / BN, (int)(Mr / BM)), blk, 0, stream>>>(
        h1h, Dm, d2h, d2_b, Dm, Dm, 0, 0, Ls, deltaf, nullptr, nullptr, nullptr);
    // xg = x + sigmoid(h2 @ a2^T + a2_b) * delta
    gemm_bf16_wmma<<<dim3(Dm / BN, (int)(Mr / BM)), blk, 0, stream>>>(
        h2h, Dm, a2h, a2_b, Dm, Dm, 2, 0, Ls, xgf, nullptr, x, deltaf);

    // layernorm -> xn (fwd) and xnf (flipped) bf16
    layernorm_kernel<<<(int)Mr, 256, 0, stream>>>(xgf, ln_w, ln_b, xnh, xnfh, Ls);

    // ---- two Mamba directions, identical pipeline ----
    for (int dir = 0; dir < 2; ++dir) {
        const unsigned short* xin = dir ? xnfh : xnh;
        // xz = xn @ in_proj^T   (M x 2048, f32)
        gemm_bf16_wmma<<<dim3(2 * Di / BN, (int)(Mr / BM)), blk, 0, stream>>>(
            xin, Dm, inph[dir], nullptr, 2 * Di, Dm, 0, 0, Ls,
            xzf, nullptr, nullptr, nullptr);
        // depthwise causal conv + silu; silu(z)
        conv_silu_kernel<<<(int)(Mr * Di / 256), 256, 0, stream>>>(
            xzf, convw[dir], convb[dir], xch, szh, Ls);
        // x_db = xc @ x_proj^T  (M x 64) -> f32 + bf16
        gemm_bf16_wmma<<<dim3(64 / BN, (int)(Mr / BM)), blk, 0, stream>>>(
            xch, Di, xph[dir], nullptr, 64, Di, 6, 0, Ls,
            xdbf, xdbh, nullptr, nullptr);
        // dt = softplus(x_db[:, :32] @ dt_proj^T + dt_b)  (M x 1024)
        gemm_bf16_wmma<<<dim3(Di / BN, (int)(Mr / BM)), blk, 0, stream>>>(
            xdbh, 64, dtph[dir], dtpb[dir], Di, 32, 3, 0, Ls,
            dtf, nullptr, nullptr, nullptr);
        // sequential SSM scan, fused y*(silu(z)) -> bf16
        ssm_scan_kernel<<<Bb * 4, 256, 0, stream>>>(
            dtf, xdbf, xch, szh, Alog[dir], Dv[dir], ybf, Ls);
        // out (+)= (y @ out_proj^T) with un-flip for dir==1
        gemm_bf16_wmma<<<dim3(Dm / BN, (int)(Mr / BM)), blk, 0, stream>>>(
            ybf, Di, outph[dir], nullptr, Dm, Di, dir ? 5 : 4, dir, Ls,
            out, nullptr, nullptr, nullptr);
    }
}
